// DebuggableTransformerEncoderLayer_75488345194520
// MI455X (gfx1250) — compile-verified, matrix operand order checked by reading, not run on hardware
//
#include <hip/hip_runtime.h>

// ---------------------------------------------------------------------------
// Transformer encoder layer for MI455X (gfx1250): wave32, WMMA f16->f32,
// TDM tensor load/store for tile staging.  B=32, S=512, D=128, H=16, FF=128.
// Outputs: x (B,S,D) then a (B,H,S,S).
// ---------------------------------------------------------------------------

#define BB 32
#define SS 512
#define DD 128
#define HH 16
#define FF 128
#define MM (BB * SS)        // 16384 rows
#define NQKV (DD * HH)      // 2048

typedef __attribute__((ext_vector_type(16))) _Float16     v16h;
typedef __attribute__((ext_vector_type(2)))  __fp16       v2fp;  // cvt_pkrtz result
typedef __attribute__((ext_vector_type(8)))  float        v8f;
typedef __attribute__((ext_vector_type(4)))  float        v4f;
typedef __attribute__((ext_vector_type(4)))  unsigned int u32x4;
typedef __attribute__((ext_vector_type(4)))  int          i32x4;
typedef __attribute__((ext_vector_type(8)))  int          i32x8;

#if defined(__has_builtin)
#if __has_builtin(__builtin_amdgcn_tensor_load_to_lds) && \
    __has_builtin(__builtin_amdgcn_s_wait_tensorcnt)
#define HAVE_TDM_LD 1
#endif
#if __has_builtin(__builtin_amdgcn_tensor_store_from_lds)
#define HAVE_TDM_ST 1
#endif
#endif
#ifndef HAVE_TDM_LD
#define HAVE_TDM_LD 0
#endif
#ifndef HAVE_TDM_ST
#define HAVE_TDM_ST 0
#endif

// --------------------------- WMMA + fragment helpers ------------------------

__device__ __forceinline__ v8f wmma_f16(v16h a, v16h b, v8f c) {
  return __builtin_amdgcn_wmma_f32_16x16x32_f16(
      false, a, false, b, (short)0, c, false, false);
}

// 16x32 f16 fragment from row-major memory (A, or pre-transposed B).
// Per lane: row = lane&15, kbase = (lane>=16)*8; two contiguous 16B runs.
__device__ __forceinline__ v16h frag_rm(const _Float16* p, int ld) {
  int lane = threadIdx.x & 31;
  const _Float16* q = p + (lane & 15) * ld + ((lane >> 4) << 3);
  union { v16h v; u32x4 u[2]; } f;
  f.u[0] = *(const u32x4*)q;
  f.u[1] = *(const u32x4*)(q + 16);
  return f.v;
}

// Pack a v8f accumulator into 8 f16 (v_cvt_pk_rtz_f16_f32 x4).
__device__ __forceinline__ u32x4 pack8_f16(v8f a) {
  union { u32x4 u; v2fp h[4]; } p;
  p.h[0] = __builtin_amdgcn_cvt_pkrtz(a[0], a[1]);
  p.h[1] = __builtin_amdgcn_cvt_pkrtz(a[2], a[3]);
  p.h[2] = __builtin_amdgcn_cvt_pkrtz(a[4], a[5]);
  p.h[3] = __builtin_amdgcn_cvt_pkrtz(a[6], a[7]);
  return p.u;
}

__device__ __forceinline__ float wave_max(float x) {
#pragma unroll
  for (int o = 16; o > 0; o >>= 1) x = fmaxf(x, __shfl_xor(x, o, 32));
  return x;
}
__device__ __forceinline__ float wave_sum(float x) {
#pragma unroll
  for (int o = 16; o > 0; o >>= 1) x += __shfl_xor(x, o, 32);
  return x;
}

// ------------------- TDM: 1-D tile DMA (ISA ch.8 descriptor) ----------------
// dcode: 0=1B, 1=2B, 2=4B element size.  nelem <= 65535.

#if HAVE_TDM_LD || HAVE_TDM_ST
__device__ __forceinline__ void tdm_desc(u32x4& g0, i32x8& g1, unsigned lds_off,
                                         unsigned long long ga, unsigned nelem,
                                         unsigned dcode) {
  g0[0] = 1u;                                        // count=1 (valid D#)
  g0[1] = lds_off;                                   // lds_addr (bytes)
  g0[2] = (unsigned)ga;                              // global_addr[31:0]
  g0[3] = (unsigned)((ga >> 32) & 0x01ffffffull)     // global_addr[56:32]
          | 0x80000000u;                             // type=2 ("image")
  g1[0] = (int)(dcode << 16);                        // data_size
  g1[1] = (int)((nelem & 0xffffu) << 16);            // tensor_dim0[15:0]
  g1[2] = (int)(((nelem >> 16) & 0xffffu) | (1u << 16)); // dim0 hi, tensor_dim1=1
  g1[3] = (int)((nelem & 0xffffu) << 16);            // tile_dim0 = nelem
  g1[4] = 0;                                         // tile_dim1/2 unused
  g1[5] = (int)nelem;                                // tensor_dim0_stride
  g1[6] = 0;
  g1[7] = 0;
}
#endif

// Stage a contiguous tile global -> LDS (TDM if available), then barrier.
__device__ __forceinline__ void stage_f16_tile(_Float16* lds,
                                               const _Float16* g, int nelem) {
#if HAVE_TDM_LD
  if ((threadIdx.x >> 5) == 0) {
    u32x4 g0; i32x8 g1;
    tdm_desc(g0, g1, (unsigned)(size_t)lds,
             (unsigned long long)(size_t)g, (unsigned)nelem, 1u);
    i32x4 z4 = {};
#if __clang_major__ >= 23
    i32x8 z8 = {};
    __builtin_amdgcn_tensor_load_to_lds(g0, g1, z4, z4, z8, 0);
#else
    __builtin_amdgcn_tensor_load_to_lds(g0, g1, z4, z4, 0);
#endif
    __builtin_amdgcn_s_wait_tensorcnt(0);
  }
#else
  const u32x4* src = (const u32x4*)g;
  u32x4* dst = (u32x4*)lds;
  int n16 = nelem >> 3;
  for (int i = threadIdx.x; i < n16; i += blockDim.x) dst[i] = src[i];
#endif
  __syncthreads();
}

// Emit a contiguous tile LDS -> global (TDM store if available).
// No explicit wait needed: S_ENDPGM performs an implicit wait-idle.
__device__ __forceinline__ void emit_tile(void* gdst, const void* lds,
                                          unsigned nelem, unsigned dcode) {
#if HAVE_TDM_ST
  if ((threadIdx.x >> 5) == 0) {
    u32x4 g0; i32x8 g1;
    tdm_desc(g0, g1, (unsigned)(size_t)lds,
             (unsigned long long)(size_t)gdst, nelem, dcode);
    i32x4 z4 = {};
#if __clang_major__ >= 23
    i32x8 z8 = {};
    __builtin_amdgcn_tensor_store_from_lds(g0, g1, z4, z4, z8, 0);
#else
    __builtin_amdgcn_tensor_store_from_lds(g0, g1, z4, z4, 0);
#endif
  }
#else
  unsigned n16 = (nelem << dcode) >> 4;
  for (unsigned i = threadIdx.x; i < n16; i += blockDim.x)
    ((u32x4*)gdst)[i] = ((const u32x4*)lds)[i];
#endif
}

// ---------------------------- elementwise convert ---------------------------

__global__ void cvt_f32_to_f16(const float* __restrict__ in,
                               _Float16* __restrict__ out, int n) {
  int i = blockIdx.x * blockDim.x + threadIdx.x;
  if (i < n) out[i] = (_Float16)in[i];
}

// Transposing convert: in is K x N row-major fp32, out is N x K row-major f16.
__global__ void cvt_transpose_f16(const float* __restrict__ in,
                                  _Float16* __restrict__ out, int K, int N) {
  int n = blockIdx.x * 32 + threadIdx.x;
  int k = blockIdx.y * 8 + threadIdx.y;
  if (n < N && k < K) out[(size_t)n * K + k] = (_Float16)in[(size_t)k * N + n];
}

// ------------------------------- QKV GEMM -----------------------------------
// grid = (M/16, NQKV/128, 3), block = 256 (8 waves, one 16-col tile each).
// A tile staged via TDM load; B = pre-transposed weights (contiguous frags).
// q,k written (B,H,S,D) via LDS tile + TDM store; v written (B,H,D,S) packed.

__global__ __launch_bounds__(256) void qkv_kernel(
    const _Float16* __restrict__ seqh,
    const _Float16* __restrict__ Wqt, const _Float16* __restrict__ Wkt,
    const _Float16* __restrict__ Wvt,
    _Float16* __restrict__ qo, _Float16* __restrict__ ko,
    _Float16* __restrict__ vT) {
  __shared__ __align__(16) _Float16 As[16 * DD];   // A tile, reused as C tile

  int z = blockIdx.z;
  const _Float16* Wt = (z == 0) ? Wqt : (z == 1) ? Wkt : Wvt;

  int mt   = blockIdx.x;
  int wave = threadIdx.x >> 5;
  int lane = threadIdx.x & 31;
  int n0   = blockIdx.y * 128 + wave * 16;

  stage_f16_tile(As, seqh + (size_t)mt * 16 * DD, 16 * DD);

  v16h af[4], bf[4];
#pragma unroll
  for (int kk = 0; kk < 4; ++kk) {
    af[kk] = frag_rm(As + kk * 32, DD);
    bf[kk] = frag_rm(Wt + (size_t)n0 * DD + kk * 32, DD);
  }
  v8f acc = {};
#pragma unroll
  for (int kk = 0; kk < 4; ++kk) acc = wmma_f16(af[kk], bf[kk], acc);

  int col = lane & 15, rb = (lane >> 4) * 8;
  int b_  = mt >> 5;                 // 32 row-tiles per batch
  int s0  = (mt * 16) & (SS - 1);
  size_t bh = (size_t)b_ * HH + blockIdx.y;   // h == blockIdx.y

  if (z == 2) {
    // V^T: per lane the 8 results are consecutive in s -> one b128 store.
    int d = wave * 16 + col;
    _Float16* dst = vT + (bh * DD + d) * SS + s0 + rb;
    *(u32x4*)dst = pack8_f16(acc);
  } else {
    // Q/K: block tile (16 x 128) is contiguous 4 KB in (B,H,S,D).
    __syncthreads();                 // As no longer needed as A tile
#pragma unroll
    for (int r = 0; r < 8; ++r)
      As[(rb + r) * DD + wave * 16 + col] = (_Float16)acc[r];
    __syncthreads();
    _Float16* dst = ((z == 0) ? qo : ko) + (bh * SS + s0) * DD;
    emit_tile(dst, As, 16 * DD, 1u);
  }
}

// ------------------------------ Attention -----------------------------------
// grid = (S/16, B*H), block = 256.  Scores 16x512 fp32 (32 KB) + f16 probs
// (16 KB) in LDS.  Writes `a` to d_out; accumulates w_o[h]*(a@v) into o via
// f32 atomics (collapses the head-mix Linear(H,1)).

__global__ __launch_bounds__(256) void attn_kernel(
    const _Float16* __restrict__ q, const _Float16* __restrict__ k,
    const _Float16* __restrict__ vT, const unsigned char* __restrict__ mask,
    const float* __restrict__ w_o, float* __restrict__ a_out,
    float* __restrict__ o_acc) {
  __shared__ __align__(16) float    sc[16 * SS];   // 32 KB raw scores
  __shared__ __align__(16) _Float16 sch[16 * SS];  // 16 KB f16 probabilities

  int qt = blockIdx.x;                             // q tile 0..31
  int bh = blockIdx.y;                             // 0..511
  int b  = bh >> 4, h = bh & 15;
  int wave = threadIdx.x >> 5;
  int lane = threadIdx.x & 31;

  const _Float16* qb  = q  + ((size_t)bh * SS + qt * 16) * DD;
  const _Float16* kb  = k  + (size_t)bh * SS * DD;
  const _Float16* vTb = vT + (size_t)bh * DD * SS;

  v16h qa[4];
#pragma unroll
  for (int i = 0; i < 4; ++i) qa[i] = frag_rm(qb + i * 32, DD);

  const float scale = 0.08838834764831845f;        // 1/sqrt(128)

  // scores: each wave computes 4 column tiles (16x16 each)
#pragma unroll
  for (int t = 0; t < 4; ++t) {
    int ct = wave * 4 + t;
    if (t < 3)
      __builtin_prefetch(kb + (size_t)(ct + 1) * 16 * DD + lane * 64, 0, 1);
    v16h bf[4];
#pragma unroll
    for (int kk = 0; kk < 4; ++kk)
      bf[kk] = frag_rm(kb + (size_t)ct * 16 * DD + kk * 32, DD);
    v8f acc = {};
#pragma unroll
    for (int kk = 0; kk < 4; ++kk) acc = wmma_f16(qa[kk], bf[kk], acc);
    int col = lane & 15, rb = (lane >> 4) * 8;
#pragma unroll
    for (int r = 0; r < 8; ++r)
      sc[(rb + r) * SS + ct * 16 + col] = acc[r] * scale;
  }
  __syncthreads();

  // masked softmax: wave handles rows 2w,2w+1; lane covers 16 CONSECUTIVE cols
#pragma unroll
  for (int rr = 0; rr < 2; ++rr) {
    int row  = wave * 2 + rr;
    int qidx = qt * 16 + row;
    bool mq = mask[b * SS + qidx] != 0;
    const float* srow         = sc   + row * SS + lane * 16;
    const unsigned char* mrow = mask + b * SS   + lane * 16;
    float vals[16];
    float mx = -3.4e38f;
#pragma unroll
    for (int i = 0; i < 16; ++i) {
      float s = srow[i] + ((mq && mrow[i]) ? 0.0f : -1.0e9f);
      vals[i] = s;
      mx = fmaxf(mx, s);
    }
    mx = wave_max(mx);
    float sum = 0.0f;
#pragma unroll
    for (int i = 0; i < 16; ++i) {
      vals[i] = __expf(vals[i] - mx);
      sum += vals[i];
    }
    sum = wave_sum(sum);
    float inv = 1.0f / sum;

    // fp32 `a` straight to global (64B contiguous per lane)
    float* arow = a_out + ((size_t)bh * SS + qidx) * SS + lane * 16;
#pragma unroll
    for (int i4 = 0; i4 < 4; ++i4) {
      v4f o;
#pragma unroll
      for (int j = 0; j < 4; ++j) o[j] = vals[i4 * 4 + j] * inv;
      *(v4f*)(arow + i4 * 4) = o;
    }
    // f16 probabilities into LDS for the AV WMMA (2x b128 ds stores)
    union { u32x4 u[2]; v2fp h[8]; } pz;
#pragma unroll
    for (int j = 0; j < 8; ++j)
      pz.h[j] = __builtin_amdgcn_cvt_pkrtz(vals[2 * j] * inv,
                                           vals[2 * j + 1] * inv);
    _Float16* hrow = sch + row * SS + lane * 16;
    *(u32x4*)hrow       = pz.u[0];
    *(u32x4*)(hrow + 8) = pz.u[1];
  }
  __syncthreads();

  // heads tile: (16x512) @ (512x128) with V^T fragments; one N-tile per wave
  float wo = w_o[h];
  int n0 = wave * 16;
  const _Float16* vrow = vTb + (size_t)n0 * SS;
  v8f acc = {};
  for (int kt0 = 0; kt0 < 32; kt0 += 4) {
    v16h af[4], bf[4];
#pragma unroll
    for (int u = 0; u < 4; ++u) {
      af[u] = frag_rm(sch + (kt0 + u) * 32, SS);    // ds b128
      bf[u] = frag_rm(vrow + (kt0 + u) * 32, SS);   // global b128
    }
#pragma unroll
    for (int u = 0; u < 4; ++u) acc = wmma_f16(af[u], bf[u], acc);
  }
  int col = lane & 15, rb = (lane >> 4) * 8;
  float* ob = o_acc + ((size_t)b * SS + qt * 16) * DD + n0;
#pragma unroll
  for (int r = 0; r < 8; ++r)
    unsafeAtomicAdd(&ob[(size_t)(rb + r) * DD + col], acc[r] * wo);
}

// ------------------------- residual + LayerNorm ------------------------------

__global__ __launch_bounds__(256) void add_ln_kernel(
    const float* __restrict__ res, const float* __restrict__ add,
    const float* __restrict__ g, const float* __restrict__ bta,
    float* __restrict__ out_f, _Float16* __restrict__ out_h) {
  int row  = blockIdx.x * 8 + (threadIdx.x >> 5);
  int lane = threadIdx.x & 31;
  size_t base = (size_t)row * DD;

  float vv[4];
  float s = 0.0f;
#pragma unroll
  for (int i = 0; i < 4; ++i) {
    int c = lane + 32 * i;
    vv[i] = res[base + c] + add[base + c];
    s += vv[i];
  }
  s = wave_sum(s);
  float mu = s * (1.0f / DD);
  float qs = 0.0f;
#pragma unroll
  for (int i = 0; i < 4; ++i) {
    float d = vv[i] - mu;
    qs += d * d;
  }
  qs = wave_sum(qs);
  float inv = rsqrtf(qs * (1.0f / DD) + 1e-5f);
#pragma unroll
  for (int i = 0; i < 4; ++i) {
    int c = lane + 32 * i;
    float o = g[c] * (vv[i] - mu) * inv + bta[c];
    out_f[base + c] = o;
    if (out_h) out_h[base + c] = (_Float16)o;
  }
}

// ------------------------------ FFN GEMM ------------------------------------
// (16384x128)@(128x128) + bias [+ relu], W pre-transposed (N x K).
// A tile staged via TDM load; C tile emitted via TDM store.

__global__ __launch_bounds__(256) void ffn_gemm_kernel(
    const _Float16* __restrict__ A, const _Float16* __restrict__ Wt,
    const float* __restrict__ bias, _Float16* __restrict__ out_h,
    float* __restrict__ out_f, int do_relu) {
  __shared__ __align__(16) _Float16 As[16 * DD];   // A tile / f16 C tile (4 KB)
  __shared__ __align__(16) float    Cs[16 * FF];   // f32 C tile (8 KB)

  int mt   = blockIdx.x;
  int wave = threadIdx.x >> 5;
  int lane = threadIdx.x & 31;
  int n0   = wave * 16;

  stage_f16_tile(As, A + (size_t)mt * 16 * DD, 16 * DD);

  v16h af[4], bf[4];
#pragma unroll
  for (int kk = 0; kk < 4; ++kk) {
    af[kk] = frag_rm(As + kk * 32, DD);
    bf[kk] = frag_rm(Wt + (size_t)n0 * DD + kk * 32, DD);
  }
  v8f acc = {};
#pragma unroll
  for (int kk = 0; kk < 4; ++kk) acc = wmma_f16(af[kk], bf[kk], acc);

  int col = lane & 15, rb = (lane >> 4) * 8;
  __syncthreads();                                 // As reuse as C tile
#pragma unroll
  for (int r = 0; r < 8; ++r) {
    int n = n0 + col;
    float val = acc[r] + bias[n];
    if (do_relu) val = fmaxf(val, 0.0f);
    if (out_h) As[(rb + r) * FF + n] = (_Float16)val;
    else       Cs[(rb + r) * FF + n] = val;
  }
  __syncthreads();
  if (out_h) emit_tile(out_h + (size_t)mt * 16 * FF, As, 16 * FF, 1u);
  else       emit_tile(out_f + (size_t)mt * 16 * FF, Cs, 16 * FF, 2u);
}

// ------------------------------ launch ---------------------------------------

extern "C" void kernel_launch(void* const* d_in, const int* in_sizes, int n_in,
                              void* d_out, int out_size, void* d_ws, size_t ws_size,
                              hipStream_t stream) {
  const float*         seq  = (const float*)d_in[0];
  const unsigned char* msk  = (const unsigned char*)d_in[1];
  const float*         Wq   = (const float*)d_in[2];
  const float*         Wk   = (const float*)d_in[3];
  const float*         Wv   = (const float*)d_in[4];
  const float*         w_o  = (const float*)d_in[5];
  const float*         g_at = (const float*)d_in[6];
  const float*         b_at = (const float*)d_in[7];
  const float*         W1   = (const float*)d_in[8];
  const float*         b1   = (const float*)d_in[9];
  const float*         W2   = (const float*)d_in[10];
  const float*         b2   = (const float*)d_in[11];
  const float*         g_ff = (const float*)d_in[12];
  const float*         b_ff = (const float*)d_in[13];

  float* out_x = (float*)d_out;                       // (B,S,D)
  float* out_a = out_x + (size_t)MM * DD;             // (B,H,S,S)

  // workspace carve-up (all chunk sizes are multiples of 256 bytes)
  char* w = (char*)d_ws;
  _Float16* seq_h = (_Float16*)w;  w += (size_t)MM * DD * 2;
  _Float16* Wqt_h = (_Float16*)w;  w += (size_t)NQKV * DD * 2;   // transposed
  _Float16* Wkt_h = (_Float16*)w;  w += (size_t)NQKV * DD * 2;
  _Float16* Wvt_h = (_Float16*)w;  w += (size_t)NQKV * DD * 2;
  _Float16* W1t_h = (_Float16*)w;  w += (size_t)FF * DD * 2;     // transposed
  _Float16* W2t_h = (_Float16*)w;  w += (size_t)DD * FF * 2;     // transposed
  _Float16* q_h   = (_Float16*)w;  w += (size_t)BB * HH * SS * DD * 2;
  _Float16* k_h   = (_Float16*)w;  w += (size_t)BB * HH * SS * DD * 2;
  _Float16* vT_h  = (_Float16*)w;  w += (size_t)BB * HH * DD * SS * 2;
  float*    o_f   = (float*)w;     w += (size_t)MM * DD * 4;
  float*    x_f   = (float*)w;     w += (size_t)MM * DD * 4;
  float*    y_f   = (float*)w;     w += (size_t)MM * DD * 4;
  _Float16* x_h   = (_Float16*)w;  w += (size_t)MM * DD * 2;
  _Float16* h_h   = (_Float16*)w;  w += (size_t)MM * FF * 2;

  // 1) staging conversions (weights transposed -> every fragment contiguous)
  cvt_f32_to_f16<<<(MM * DD + 255) / 256, 256, 0, stream>>>(seq, seq_h, MM * DD);
  {
    dim3 blk(32, 8);
    cvt_transpose_f16<<<dim3(NQKV / 32, DD / 8), blk, 0, stream>>>(Wq, Wqt_h, DD, NQKV);
    cvt_transpose_f16<<<dim3(NQKV / 32, DD / 8), blk, 0, stream>>>(Wk, Wkt_h, DD, NQKV);
    cvt_transpose_f16<<<dim3(NQKV / 32, DD / 8), blk, 0, stream>>>(Wv, Wvt_h, DD, NQKV);
    cvt_transpose_f16<<<dim3(FF / 32, DD / 8),   blk, 0, stream>>>(W1, W1t_h, DD, FF);
    cvt_transpose_f16<<<dim3(DD / 32, FF / 8),   blk, 0, stream>>>(W2, W2t_h, FF, DD);
  }

  // 2) zero the head-mix accumulator
  (void)hipMemsetAsync(o_f, 0, (size_t)MM * DD * sizeof(float), stream);

  // 3) QKV projections -> q,k (B,H,S,D) f16; v transposed (B,H,D,S)
  qkv_kernel<<<dim3(MM / 16, NQKV / 128, 3), 256, 0, stream>>>(
      seq_h, Wqt_h, Wkt_h, Wvt_h, q_h, k_h, vT_h);

  // 4) attention: scores, softmax, `a` to d_out, w_o-weighted a@v into o_f
  attn_kernel<<<dim3(SS / 16, BB * HH), 256, 0, stream>>>(
      q_h, k_h, vT_h, msk, w_o, out_a, o_f);

  // 5) x = LN(seq + o)
  add_ln_kernel<<<MM / 8, 256, 0, stream>>>(seq, o_f, g_at, b_at, x_f, x_h);

  // 6) FFN
  ffn_gemm_kernel<<<MM / 16, 256, 0, stream>>>(x_h, W1t_h, b1, h_h, nullptr, 1);
  ffn_gemm_kernel<<<MM / 16, 256, 0, stream>>>(h_h, W2t_h, b2, nullptr, y_f, 0);

  // 7) final LN -> d_out x section
  add_ln_kernel<<<MM / 8, 256, 0, stream>>>(x_f, y_f, g_ff, b_ff, out_x, nullptr);
}
